// Ensembler_39737037423019
// MI455X (gfx1250) — compile-verified, matrix operand order checked by reading, not run on hardware
//
#include <hip/hip_runtime.h>
#include <hip/hip_bf16.h>

// Problem constants (fixed by the reference: S=3, Q=64, X=128, Y=128, Z=32, C=21)
#define QN    64
#define NVOX  524288          // 128*128*32
#define NC    21
#define KBLK  256             // K bytes staged to LDS per block per chunk
#define GRIDW 256             // blocks in the WMMA kernel
#define NCHUNK (NVOX / (GRIDW * KBLK))   // 8
#define IOU_TH 0.2f

typedef int v8i __attribute__((ext_vector_type(8)));

__device__ __forceinline__ float sigmoidf(float x) {
    return 1.0f / (1.0f + __expf(-x));
}

__device__ __forceinline__ unsigned int pack4(const float4 f, float th, int &cnt) {
    unsigned int b0 = f.x > th;
    unsigned int b1 = f.y > th;
    unsigned int b2 = f.z > th;
    unsigned int b3 = f.w > th;
    cnt = (int)(b0 + b1 + b2 + b3);
    return b0 | (b1 << 8) | (b2 << 16) | (b3 << 24);   // K ascending in bytes [7:0]..[31:24]
}

// ---------------------------------------------------------------------------
// occ[n] = (argmax_c sem[c,n]) != 0   (first-max semantics via strict >)
// ---------------------------------------------------------------------------
__global__ void occ_kernel(const float* __restrict__ sem, unsigned char* __restrict__ occ) {
    int n = blockIdx.x * blockDim.x + threadIdx.x;
    if (n >= NVOX) return;
    float best = sem[n];
    int bc = 0;
    for (int c = 1; c < NC; ++c) {
        float v = sem[(size_t)c * NVOX + n];
        if (v > best) { best = v; bc = c; }
    }
    occ[n] = (bc != 0) ? 1 : 0;
}

// ---------------------------------------------------------------------------
// zero inter[64*64] + sumA[64] + sumB[64]
// ---------------------------------------------------------------------------
__global__ void zero_kernel(int* __restrict__ p, int n) {
    int i = blockIdx.x * blockDim.x + threadIdx.x;
    if (i < n) p[i] = 0;
}

// ---------------------------------------------------------------------------
// inter[q][p] = sum_n (A[q,n] > thA) * (B[p,n] > thB)  via V_WMMA_I32_16X16X64_IU8
// sumA[q] = sum_n (A[q,n] > thA), sumB[p] likewise.
// 512 threads = 16 waves; wave (qt,pt) owns a 16x16 tile of inter.
// LDS: 128 rows (64 A + 64 B) x KBLK packed 0/1 bytes.
// Staging is fully unrolled (8 batched b128 loads per matrix per thread) so
// loads issue back-to-back instead of load->wait->load.
// ---------------------------------------------------------------------------
__global__ __launch_bounds__(512) void inter_wmma_kernel(
    const float* __restrict__ A, const float* __restrict__ B,
    float thA, float thB,
    int* __restrict__ inter, int* __restrict__ sumA, int* __restrict__ sumB)
{
    __shared__ unsigned char ldsM[128 * KBLK];   // 32 KB
    __shared__ int ldsSum[128];

    const int tid  = threadIdx.x;
    const int wave = tid >> 5;
    const int lane = tid & 31;
    const int qt   = wave >> 2;       // 0..3
    const int pt   = wave & 3;        // 0..3
    const int r    = lane & 15;
    const int half = lane >> 4;       // 0 or 1

    const int srow = tid >> 6;        // staging base row 0..7
    const int g    = tid & 63;        // staging float4 group (column) 0..63

    if (tid < 128) ldsSum[tid] = 0;
    __syncthreads();

    v8i acc = {0, 0, 0, 0, 0, 0, 0, 0};

    for (int chunk = 0; chunk < NCHUNK; ++chunk) {
        const size_t kbase = ((size_t)chunk * GRIDW + blockIdx.x) * KBLK;

        // ---- stage A: 8 batched loads, then pack -> LDS + row popcounts ----
        {
            float4 f[8];
            #pragma unroll
            for (int k = 0; k < 8; ++k) {
                const int row = srow + k * 8;
                f[k] = *(const float4*)(A + (size_t)row * NVOX + kbase + g * 4);
            }
            #pragma unroll
            for (int k = 0; k < 8; ++k) {
                const int row = srow + k * 8;
                int c;
                const unsigned int m = pack4(f[k], thA, c);
                *(unsigned int*)&ldsM[row * KBLK + g * 4] = m;
                if (c) atomicAdd(&ldsSum[row], c);
            }
        }
        // ---- stage B ----
        {
            float4 f[8];
            #pragma unroll
            for (int k = 0; k < 8; ++k) {
                const int row = srow + k * 8;
                f[k] = *(const float4*)(B + (size_t)row * NVOX + kbase + g * 4);
            }
            #pragma unroll
            for (int k = 0; k < 8; ++k) {
                const int row = srow + k * 8;
                int c;
                const unsigned int m = pack4(f[k], thB, c);
                *(unsigned int*)&ldsM[(64 + row) * KBLK + g * 4] = m;
                if (c) atomicAdd(&ldsSum[64 + row], c);
            }
        }
        __syncthreads();

        // ---- WMMA over this slab: KBLK/64 = 4 steps ----
        const unsigned char* rowA = &ldsM[(qt * 16 + r) * KBLK];
        const unsigned char* rowB = &ldsM[(64 + pt * 16 + r) * KBLK];
        #pragma unroll
        for (int ks = 0; ks < KBLK; ks += 64) {
            v8i a, b;
            // 8-bit A fragment layout: lane<16 -> K {0-7,16-23,32-39,48-55}; lane>=16 -> +8
            #pragma unroll
            for (int j = 0; j < 4; ++j) {
                const int* pa = (const int*)(rowA + ks + j * 16 + half * 8);
                a[2 * j]     = pa[0];
                a[2 * j + 1] = pa[1];
                const int* pb = (const int*)(rowB + ks + j * 16 + half * 8);
                b[2 * j]     = pb[0];
                b[2 * j + 1] = pb[1];
            }
            acc = __builtin_amdgcn_wmma_i32_16x16x64_iu8(false, a, false, b, acc, false, false);
        }
        __syncthreads();
    }

    // ---- writeback: C layout -> inter element (M = v + 8*half, N = lane&15) ----
    #pragma unroll
    for (int v = 0; v < 8; ++v) {
        const int m = qt * 16 + v + 8 * half;
        const int n = pt * 16 + (lane & 15);
        atomicAdd(&inter[m * QN + n], acc[v]);
    }
    if (tid < 128) {
        if (tid < 64) atomicAdd(&sumA[tid], ldsSum[tid]);
        else          atomicAdd(&sumB[tid - 64], ldsSum[tid]);
    }
}

// ---------------------------------------------------------------------------
// Per-anchor-query IoU row max/argmax; pass2 also computes keep.
// 1 block x 64 threads.
// ---------------------------------------------------------------------------
__global__ void iou_kernel(const int* __restrict__ inter,
                           const int* __restrict__ sumA, const int* __restrict__ sumB,
                           float* __restrict__ iouOut, int* __restrict__ idxOut,
                           int* __restrict__ matchedOut,
                           const float* __restrict__ iouPrev, float* __restrict__ keepOut)
{
    const int q = threadIdx.x;
    if (q >= QN) return;
    const float sa = (float)sumA[q];
    float best = -1.0f;
    int bidx = 0;
    for (int p = 0; p < QN; ++p) {
        const int iv = inter[q * QN + p];
        const float un = sa + (float)sumB[p] - (float)iv;
        const float iou = (float)iv / fmaxf(un, 1.0f);
        if (iou > best) { best = iou; bidx = p; }   // first max (strict >)
    }
    iouOut[q]     = best;
    idxOut[q]     = bidx;
    matchedOut[q] = (best > IOU_TH) ? 1 : 0;
    if (keepOut) {
        const float mean = 0.5f * (iouPrev[q] + best);
        keepOut[q] = (mean > IOU_TH) ? 1.0f : 0.0f;
    }
}

// ---------------------------------------------------------------------------
// anchor_v = matched1[q] ? (sigmoid(l0)+sigmoid(l1[idx1[q]]))/2 : sigmoid(l0)
// written to d_out (Q x N).  4 floats per thread.
// ---------------------------------------------------------------------------
__global__ void merge1_kernel(const float* __restrict__ l0, const float* __restrict__ l1,
                              const int* __restrict__ idx1, const int* __restrict__ matched1,
                              float* __restrict__ outv)
{
    const long long gid = (long long)blockIdx.x * blockDim.x + threadIdx.x;
    const int q  = (int)(gid >> 17);              // NVOX/4 = 131072 = 2^17 float4 groups per row
    const long long n4 = gid & 131071;
    const size_t off = (size_t)q * NVOX + n4 * 4;
    float4 a = *(const float4*)(l0 + off);
    a.x = sigmoidf(a.x); a.y = sigmoidf(a.y); a.z = sigmoidf(a.z); a.w = sigmoidf(a.w);
    if (matched1[q]) {
        const size_t offb = (size_t)idx1[q] * NVOX + n4 * 4;
        float4 b = *(const float4*)(l1 + offb);
        a.x = 0.5f * (a.x + sigmoidf(b.x));
        a.y = 0.5f * (a.y + sigmoidf(b.y));
        a.z = 0.5f * (a.z + sigmoidf(b.z));
        a.w = 0.5f * (a.w + sigmoidf(b.w));
    }
    *(float4*)(outv + off) = a;
}

// ---------------------------------------------------------------------------
// out = (matched2 ? (anchor*2 + sigmoid(l2[idx2]))/3 : anchor) * keep[q] * occ[n]
// in-place on d_out.
// ---------------------------------------------------------------------------
__global__ void final_kernel(float* __restrict__ outv, const float* __restrict__ l2,
                             const int* __restrict__ idx2, const int* __restrict__ matched2,
                             const float* __restrict__ keep, const unsigned char* __restrict__ occ)
{
    const long long gid = (long long)blockIdx.x * blockDim.x + threadIdx.x;
    const int q  = (int)(gid >> 17);
    const long long n4 = gid & 131071;
    const size_t off = (size_t)q * NVOX + n4 * 4;
    float4 a = *(const float4*)(outv + off);
    if (matched2[q]) {
        const size_t offb = (size_t)idx2[q] * NVOX + n4 * 4;
        float4 b = *(const float4*)(l2 + offb);
        const float inv3 = 1.0f / 3.0f;
        a.x = (a.x * 2.0f + sigmoidf(b.x)) * inv3;
        a.y = (a.y * 2.0f + sigmoidf(b.y)) * inv3;
        a.z = (a.z * 2.0f + sigmoidf(b.z)) * inv3;
        a.w = (a.w * 2.0f + sigmoidf(b.w)) * inv3;
    }
    const float k = keep[q];
    const unsigned int oc = *(const unsigned int*)(occ + n4 * 4);
    a.x *= k * (float)(oc & 0xff);
    a.y *= k * (float)((oc >> 8) & 0xff);
    a.z *= k * (float)((oc >> 16) & 0xff);
    a.w *= k * (float)((oc >> 24) & 0xff);
    *(float4*)(outv + off) = a;
}

// ---------------------------------------------------------------------------
extern "C" void kernel_launch(void* const* d_in, const int* in_sizes, int n_in,
                              void* d_out, int out_size, void* d_ws, size_t ws_size,
                              hipStream_t stream) {
    const float* vox = (const float*)d_in[0];             // [3, 64, N]
    // d_in[1] (query_logits) is dead code in the reference output: ignored.
    const float* sem = (const float*)d_in[2];             // [21, N]
    float* outv = (float*)d_out;                          // [64, N]

    const float* l0 = vox;
    const float* l1 = vox + (size_t)QN * NVOX;
    const float* l2 = vox + 2 * (size_t)QN * NVOX;

    // workspace carve-up (ints unless noted)
    int*   inter    = (int*)d_ws;                 // 64*64
    int*   sumA     = inter + QN * QN;            // 64
    int*   sumB     = sumA + QN;                  // 64
    int*   idx1     = sumB + QN;                  // 64
    int*   matched1 = idx1 + QN;                  // 64
    int*   idx2     = matched1 + QN;              // 64
    int*   matched2 = idx2 + QN;                  // 64
    float* iou1     = (float*)(matched2 + QN);    // 64
    float* iou2     = iou1 + QN;                  // 64
    float* keep     = iou2 + QN;                  // 64
    unsigned char* occ = (unsigned char*)(keep + QN);   // N bytes

    const int ZERON = QN * QN + 2 * QN;           // inter + sumA + sumB

    // 1) occupancy mask from semantic argmax
    occ_kernel<<<NVOX / 256, 256, 0, stream>>>(sem, occ);

    // 2) pass 1: inter(mask(l0>0), mask(l1>0))
    zero_kernel<<<(ZERON + 255) / 256, 256, 0, stream>>>(inter, ZERON);
    inter_wmma_kernel<<<GRIDW, 512, 0, stream>>>(l0, l1, 0.0f, 0.0f, inter, sumA, sumB);
    iou_kernel<<<1, 64, 0, stream>>>(inter, sumA, sumB, iou1, idx1, matched1, nullptr, nullptr);

    // 3) merged anchor probabilities -> d_out
    merge1_kernel<<<(QN * NVOX / 4) / 256, 256, 0, stream>>>(l0, l1, idx1, matched1, outv);

    // 4) pass 2: inter(mask(anchor>0.5), mask(l2>0))
    zero_kernel<<<(ZERON + 255) / 256, 256, 0, stream>>>(inter, ZERON);
    inter_wmma_kernel<<<GRIDW, 512, 0, stream>>>(outv, l2, 0.5f, 0.0f, inter, sumA, sumB);
    iou_kernel<<<1, 64, 0, stream>>>(inter, sumA, sumB, iou2, idx2, matched2, iou1, keep);

    // 5) final merge * keep * occ, in place on d_out
    final_kernel<<<(QN * NVOX / 4) / 256, 256, 0, stream>>>(outv, l2, idx2, matched2, keep, occ);
}
